// NN_Attention_12747462934738
// MI455X (gfx1250) — compile-verified, matrix operand order checked by reading, not run on hardware
//
#include <hip/hip_runtime.h>
#include <hip/hip_bf16.h>

// ---------------------------------------------------------------------------
// Additive (Bahdanau) attention, fused per (batch, head) workgroup.
// GEMM Uk = K @ Uw^T done with v_wmma_f32_16x16x32_bf16 (bf16 in, f32 acc).
// k-loop unroll is capped at 4 to keep VGPR pressure < 256 (round-1 build
// fully unrolled, used the VGPR-MSB path and spilled to scratch).
// ---------------------------------------------------------------------------

typedef __attribute__((ext_vector_type(16))) __bf16 v16bf;
typedef __attribute__((ext_vector_type(8)))  float  v8f;

#define B_    64
#define L_    2048
#define D_    512
#define H_    8
#define DH_   64

// LDS layout (bytes). A: 256 rows x 520 halfs (8-half pad -> conflict-free
// b128 fragment reads), B slice: 32 rows x 520 halfs.
#define A_STRIDE   520
#define OFF_A      0
#define SZ_A       (256 * A_STRIDE * 2)        // 266240
#define OFF_B      (OFF_A + SZ_A)              // 266240
#define SZ_B       (32 * A_STRIDE * 2)         // 33280
#define OFF_SC     (OFF_B + SZ_B)              // 299520
#define SZ_SC      (2048 * 4)                  // 8192
#define OFF_UB     (OFF_SC + SZ_SC)            // 307712
#define OFF_WQ     (OFF_UB + 512 * 4)          // 309760
#define OFF_VW     (OFF_WQ + 64 * 4)           // 310016
#define OFF_RED    (OFF_VW + 64 * 4)           // 310272
#define OFF_PART   (OFF_RED + 64)              // 310336
#define SMEM_BYTES (OFF_PART + 256 * 4)        // 311360  (<= 320 KB WGP LDS)

union Frag {
    uint4 u[2];
    v16bf v;
};

__device__ __forceinline__ unsigned int bf16u(float f) {
    unsigned int u = __float_as_uint(f);
    u += 0x7fffu + ((u >> 16) & 1u);           // round-to-nearest-even
    return u >> 16;
}
__device__ __forceinline__ unsigned int pack2(float a, float b) {
    return bf16u(a) | (bf16u(b) << 16);
}

__device__ __forceinline__ v8f wmma_bf16(const Frag& a, const Frag& b, v8f c) {
    return __builtin_amdgcn_wmma_f32_16x16x32_bf16(
        /*neg_a=*/false, a.v, /*neg_b=*/false, b.v,
        /*c_mod=*/(short)0, c, /*reuse_a=*/false, /*reuse_b=*/false);
}

// Reduce a 16x16 f32 C-tile into the score array:
//   score[ll] += tanh(wq[c&63] + C[r][c] + ub[c]) * vw[c&63],  ll = r*8 + c>>6
// All 16 cols of a tile share one 64-block -> single ll per row.
__device__ __forceinline__ void score_tile(v8f acc, int rowbase, int ntbase,
                                           int lane, const float* wqs,
                                           const float* ubs, const float* vws,
                                           float* sc) {
    const int nl  = lane & 15;
    const int c   = ntbase + nl;
    const float wqv = wqs[c & 63];
    const float ubv = ubs[c];
    const float vwv = vws[c & 63];
    const int j  = ntbase >> 6;
    const int rb = rowbase + (lane >> 4) * 8;   // C layout: VGPR v -> M = v + (l>>4)*8
#pragma unroll
    for (int v = 0; v < 8; ++v) {
        float t = tanhf(acc[v] + ubv + wqv) * vwv;
        t += __shfl_xor(t, 1, 16);
        t += __shfl_xor(t, 2, 16);
        t += __shfl_xor(t, 4, 16);
        t += __shfl_xor(t, 8, 16);
        if (nl == 0) atomicAdd(&sc[(rb + v) * 8 + j], t);   // ds_add_f32
    }
}

__global__ void __launch_bounds__(256)
wq_gemv_kernel(const float* __restrict__ Q, const float* __restrict__ Ww,
               const float* __restrict__ Wb, float* __restrict__ wqout) {
    __shared__ float qs[D_];
    const int b = blockIdx.x, tid = threadIdx.x;
    qs[tid]       = Q[(size_t)b * D_ + tid];
    qs[tid + 256] = Q[(size_t)b * D_ + 256 + tid];
    __syncthreads();
#pragma unroll
    for (int dd = 0; dd < 2; ++dd) {
        const int d = tid + dd * 256;
        const float4* wr = (const float4*)(Ww + (size_t)d * D_);
        float acc = Wb[d];
#pragma clang loop unroll_count(8)
        for (int k4 = 0; k4 < D_ / 4; ++k4) {
            float4 w = wr[k4];
            acc += w.x * qs[k4 * 4 + 0] + w.y * qs[k4 * 4 + 1] +
                   w.z * qs[k4 * 4 + 2] + w.w * qs[k4 * 4 + 3];
        }
        wqout[(size_t)b * D_ + d] = acc;
    }
}

__global__ void __launch_bounds__(256)
fused_attn_kernel(const float* __restrict__ K,  const float* __restrict__ Uw,
                  const float* __restrict__ Ub, const float* __restrict__ Vw,
                  const float* __restrict__ Vb, const float* __restrict__ Wq,
                  float* __restrict__ out) {
    extern __shared__ char smem[];
    unsigned short* Ald = (unsigned short*)(smem + OFF_A);
    unsigned short* Bld = (unsigned short*)(smem + OFF_B);
    float* sc   = (float*)(smem + OFF_SC);
    float* ubs  = (float*)(smem + OFF_UB);
    float* wqs  = (float*)(smem + OFF_WQ);
    float* vws  = (float*)(smem + OFF_VW);
    float* red  = (float*)(smem + OFF_RED);
    float* part = (float*)(smem + OFF_PART);

    const int tid  = threadIdx.x;
    const int wave = tid >> 5;
    const int lane = tid & 31;
    const int b = blockIdx.x >> 3;
    const int h = blockIdx.x & 7;

    // Head h's flat reshape == contiguous 256x512 row block of K[b].
    const float* Kp = K + ((size_t)b * L_ + (size_t)h * 256) * D_;

    // ---- stage biases / Wq / Vw, zero score array ----
    if (tid < 128) ((float4*)ubs)[tid] = ((const float4*)Ub)[tid];
    if (tid < 16) {
        ((float4*)vws)[tid] = ((const float4*)Vw)[tid];
        ((float4*)wqs)[tid] = ((const float4*)(Wq + (size_t)b * D_ + h * DH_))[tid];
    }
    for (int i = tid; i < 2048; i += 256) sc[i] = 0.f;

    // ---- stage A = K block (256x512) as bf16 in LDS ----
#pragma clang loop unroll_count(8)
    for (int i = 0; i < 128; ++i) {
        const int idx = i * 256 + tid;          // float4 index
        const int f = idx * 4;
        const int r = f >> 9, c = f & 511;
        float4 v = ((const float4*)Kp)[idx];
        uint2 p;
        p.x = pack2(v.x, v.y);
        p.y = pack2(v.z, v.w);
        *(uint2*)(Ald + r * A_STRIDE + c) = p;
    }
    __syncthreads();

    const int mrow  = wave * 32;                // each wave owns 32 M rows
    const int lrow  = lane & 15;
    const int khalf = lane >> 4;

    // ---- 16 passes over N (32 cols each): GEMM + fused tanh/Vw reduction ----
#pragma clang loop unroll_count(1)
    for (int pass = 0; pass < 16; ++pass) {
        // stage B slice: Uw rows [pass*32, +32) (== Uw^T cols), bf16 [n][k]
#pragma clang loop unroll_count(4)
        for (int i = 0; i < 16; ++i) {
            const int idx = i * 256 + tid;
            const int f = idx * 4;
            const int n = f >> 9, c = f & 511;
            float4 v = ((const float4*)(Uw + (size_t)pass * 32 * D_))[idx];
            uint2 p;
            p.x = pack2(v.x, v.y);
            p.y = pack2(v.z, v.w);
            *(uint2*)(Bld + n * A_STRIDE + c) = p;
        }
        __syncthreads();

        v8f a00 = {}, a01 = {}, a10 = {}, a11 = {};
        // Unroll capped at 4: 16 WMMAs + 32 ds_load_b128 per region keeps
        // in-flight fragments at ~128 VGPRs -> no MSB addressing, no spills.
#pragma clang loop unroll_count(4)
        for (int ks = 0; ks < 16; ++ks) {
            const int kb = ks * 32 + khalf * 8; // A: low 8 halfs K=kh*8.., high 8 K=16+kh*8..
            Frag fa0, fa1, fb0, fb1;
            const unsigned short* ap0 = Ald + (mrow + lrow) * A_STRIDE + kb;
            fa0.u[0] = *(const uint4*)(ap0);
            fa0.u[1] = *(const uint4*)(ap0 + 16);
            const unsigned short* ap1 = ap0 + 16 * A_STRIDE;
            fa1.u[0] = *(const uint4*)(ap1);
            fa1.u[1] = *(const uint4*)(ap1 + 16);
            const unsigned short* bp0 = Bld + lrow * A_STRIDE + kb;
            fb0.u[0] = *(const uint4*)(bp0);
            fb0.u[1] = *(const uint4*)(bp0 + 16);
            const unsigned short* bp1 = bp0 + 16 * A_STRIDE;
            fb1.u[0] = *(const uint4*)(bp1);
            fb1.u[1] = *(const uint4*)(bp1 + 16);
            a00 = wmma_bf16(fa0, fb0, a00);
            a01 = wmma_bf16(fa0, fb1, a01);
            a10 = wmma_bf16(fa1, fb0, a10);
            a11 = wmma_bf16(fa1, fb1, a11);
        }
        const int nt0 = pass * 32, nt1 = pass * 32 + 16;
        score_tile(a00, mrow,      nt0, lane, wqs, ubs, vws, sc);
        score_tile(a01, mrow,      nt1, lane, wqs, ubs, vws, sc);
        score_tile(a10, mrow + 16, nt0, lane, wqs, ubs, vws, sc);
        score_tile(a11, mrow + 16, nt1, lane, wqs, ubs, vws, sc);
        __syncthreads();
    }

    // ---- softmax over the 2048 scores (Vb is shift-invariant but included) ----
    const float vb = Vb[0];
    float s[8];
    {
        float4 s0 = ((float4*)sc)[tid * 2], s1 = ((float4*)sc)[tid * 2 + 1];
        s[0] = s0.x + vb; s[1] = s0.y + vb; s[2] = s0.z + vb; s[3] = s0.w + vb;
        s[4] = s1.x + vb; s[5] = s1.y + vb; s[6] = s1.z + vb; s[7] = s1.w + vb;
    }
    float m = s[0];
#pragma unroll
    for (int i = 1; i < 8; ++i) m = fmaxf(m, s[i]);
#pragma unroll
    for (int mask = 1; mask < 32; mask <<= 1) m = fmaxf(m, __shfl_xor(m, mask, 32));
    if (lane == 0) red[wave] = m;
    __syncthreads();
    float gm = red[0];
#pragma unroll
    for (int i = 1; i < 8; ++i) gm = fmaxf(gm, red[i]);
    __syncthreads();
    float sum = 0.f;
#pragma unroll
    for (int i = 0; i < 8; ++i) { s[i] = __expf(s[i] - gm); sum += s[i]; }
#pragma unroll
    for (int mask = 1; mask < 32; mask <<= 1) sum += __shfl_xor(sum, mask, 32);
    if (lane == 0) red[wave] = sum;
    __syncthreads();
    float gs = red[0];
#pragma unroll
    for (int i = 1; i < 8; ++i) gs += red[i];
    const float inv = 1.f / gs;

    // write dist (out[32768 ..]) and keep it in LDS for the attn pass
    float* distOut = out + (size_t)B_ * D_ + ((size_t)(b * H_ + h)) * L_;
    float4 d0, d1;
    d0.x = s[0] * inv; d0.y = s[1] * inv; d0.z = s[2] * inv; d0.w = s[3] * inv;
    d1.x = s[4] * inv; d1.y = s[5] * inv; d1.z = s[6] * inv; d1.w = s[7] * inv;
    ((float4*)distOut)[tid * 2]     = d0;
    ((float4*)distOut)[tid * 2 + 1] = d1;
    ((float4*)sc)[tid * 2]     = d0;
    ((float4*)sc)[tid * 2 + 1] = d1;
    __syncthreads();

    // ---- attn[b, h*64+dh] = sum_{r,j} dist[r*8+j] * Kp[r*512 + j*64 + dh] ----
    {
        const int dh = tid & 63, g = tid >> 6;
        float acc = 0.f;
#pragma clang loop unroll_count(2)
        for (int r = g; r < 256; r += 4) {
            const float* row = Kp + (size_t)r * D_;
            __builtin_prefetch(row + 4 * D_, 0, 1);   // global_prefetch_b8
#pragma unroll
            for (int j = 0; j < 8; ++j)
                acc += row[j * 64 + dh] * sc[r * 8 + j];
        }
        part[tid] = acc;
    }
    __syncthreads();
    if (tid < 64) {
        float a = part[tid] + part[tid + 64] + part[tid + 128] + part[tid + 192];
        out[(size_t)b * D_ + h * DH_ + tid] = a;
    }
}

extern "C" void kernel_launch(void* const* d_in, const int* in_sizes, int n_in,
                              void* d_out, int out_size, void* d_ws, size_t ws_size,
                              hipStream_t stream) {
    const float* Q  = (const float*)d_in[0];
    const float* K  = (const float*)d_in[1];
    const float* Ww = (const float*)d_in[2];
    const float* Wb = (const float*)d_in[3];
    const float* Uw = (const float*)d_in[4];
    const float* Ub = (const float*)d_in[5];
    const float* Vw = (const float*)d_in[6];
    const float* Vb = (const float*)d_in[7];
    float* out = (float*)d_out;
    float* wq  = (float*)d_ws;            // 64*512 floats = 128 KB scratch

    wq_gemv_kernel<<<B_, 256, 0, stream>>>(Q, Ww, Wb, wq);

    hipFuncSetAttribute((const void*)fused_attn_kernel,
                        hipFuncAttributeMaxDynamicSharedMemorySize, SMEM_BYTES);
    fused_attn_kernel<<<B_ * H_, 256, SMEM_BYTES, stream>>>(K, Uw, Ub, Vw, Vb, wq, out);
}
